// WindowJointAttention_72499047956645
// MI455X (gfx1250) — compile-verified, matrix operand order checked by reading, not run on hardware
//
#include <hip/hip_runtime.h>

#define DIMC   128
#define HEADS  4
#define HD     32
#define NTOK   64
#define QKSCALE 0.17677669529663687f   // 32^-0.5

// Swizzled f16 weight buffer layout in d_ws (units: halves):
//   [0      ) wq     8 ntiles
//   [16384  ) wkv   16 ntiles (K cols 0..127 -> nt 0..7, V cols 128..255 -> nt 8..15)
//   [49152  ) wq_f   8 ntiles
//   [65536  ) wkv_f  8 ntiles (K half only, src ld = 256)
//   [81920  ) wproj  8 ntiles
//   total 98304 halves = 192KB
#define WS_WQ    0
#define WS_WKV   16384
#define WS_WQF   49152
#define WS_WKVF  65536
#define WS_WPROJ 81920
#define WS_TOTAL 98304

typedef _Float16 v16h __attribute__((ext_vector_type(16)));
typedef _Float16 v4h  __attribute__((ext_vector_type(4)));
typedef float    v8f  __attribute__((ext_vector_type(8)));

// K index inside a 16x32 f16 A/B fragment for VGPR-pair p (0..7), lane-half g (0/1).
// Per CDNA5 ISA 7.12.2: lanes 0-15 hold K=0..7 then 16..23; lanes 16-31 hold 8..15 then 24..31.
static __device__ __forceinline__ int frag_k(int p, int g) {
    return ((p < 4) ? (2 * p) : (16 + 2 * (p - 4))) + 8 * g;
}

// A-fragment (or B-fragment whose transpose is row-major in LDS): element (r, k)
// read from base[(row0 + r)*ld + col0 + k], r = lane&15.
static __device__ __forceinline__ v16h load_frag_rowmajor(const _Float16* base,
                                                          int row0, int col0, int ld) {
    const int lane = threadIdx.x & 31;
    const int r = lane & 15, g = lane >> 4;
    const _Float16* rp = base + (row0 + r) * ld + col0;
    v16h v;
#pragma unroll
    for (int p = 0; p < 8; ++p) {
        const int k = frag_k(p, g);
        v[2 * p]     = rp[k];
        v[2 * p + 1] = rp[k + 1];
    }
    return v;
}

// B-fragment whose source is row-major over K: element (k, n) from base[(k0+k)*ld + col0 + n].
static __device__ __forceinline__ v16h load_fragB_colsrc(const _Float16* base,
                                                         int k0, int col0, int ld) {
    const int lane = threadIdx.x & 31;
    const int n = lane & 15, g = lane >> 4;
    v16h v;
#pragma unroll
    for (int p = 0; p < 8; ++p) {
        const int k = frag_k(p, g);
        v[2 * p]     = base[(k0 + k) * ld + col0 + n];
        v[2 * p + 1] = base[(k0 + k + 1) * ld + col0 + n];
    }
    return v;
}

// B-fragment from the pre-swizzled f16 weight buffer: one contiguous 32B load per lane.
static __device__ __forceinline__ v16h load_fragB_swz(const _Float16* __restrict__ wsbase,
                                                      int nt, int kt) {
    const v16h* p = (const v16h*)wsbase;
    return p[(nt * 4 + kt) * 32 + (threadIdx.x & 31)];
}

static __device__ __forceinline__ v8f wmma_f16(v16h a, v16h b, v8f c) {
    // (neg_a, A, neg_b, B, c_mod, C, reuse_a, reuse_b)
    return __builtin_amdgcn_wmma_f32_16x16x32_f16(false, a, false, b, (short)0, c,
                                                  false, false);
}

// ---- Prep: convert fp32 weights -> f16, laid out in WMMA B-fragment register order.
// Element (matrix, nt, kt, lane, i): value = W[kt*32 + frag_k(i>>1, lane>>4) + (i&1)]
//                                             [nt*16 + (lane&15)]
__global__ __launch_bounds__(256)
void swizzle_weights_kernel(const float* __restrict__ wq, const float* __restrict__ wkv,
                            const float* __restrict__ wq_f, const float* __restrict__ wkv_f,
                            const float* __restrict__ wproj, _Float16* __restrict__ ws) {
    const int idx = blockIdx.x * 256 + threadIdx.x;
    if (idx >= WS_TOTAL) return;
    const float* W; int srcld, base;
    if (idx < WS_WKV)        { W = wq;    srcld = 128; base = WS_WQ; }
    else if (idx < WS_WQF)   { W = wkv;   srcld = 256; base = WS_WKV; }
    else if (idx < WS_WKVF)  { W = wq_f;  srcld = 128; base = WS_WQF; }
    else if (idx < WS_WPROJ) { W = wkv_f; srcld = 256; base = WS_WKVF; }
    else                     { W = wproj; srcld = 128; base = WS_WPROJ; }
    const int local = idx - base;
    const int i    = local & 15;
    const int lane = (local >> 4) & 31;
    const int kt   = (local >> 9) & 3;
    const int nt   = local >> 11;
    const int k = kt * 32 + frag_k(i >> 1, lane >> 4) + (i & 1);
    const int n = nt * 16 + (lane & 15);
    ws[idx] = (_Float16)W[(size_t)k * srcld + n];
}

__global__ __launch_bounds__(256)
void win_joint_attn_kernel(const float* __restrict__ x, const float* __restrict__ f,
                           const float* __restrict__ bq,  const float* __restrict__ bkv,
                           const float* __restrict__ bq_f, const float* __restrict__ bkv_f,
                           const float* __restrict__ bias_table,
                           const float* __restrict__ bproj,
                           const _Float16* __restrict__ ws,
                           float* __restrict__ out) {
    extern __shared__ unsigned char smem[];
    // Region A (32KB): x/f staging (f16), later reused for score/prob matrix SP [4][64][64] f16.
    _Float16* xh = (_Float16*)smem;                 // [64][128]
    _Float16* fh = xh + NTOK * DIMC;                // [64][128]
    _Float16* SP = (_Float16*)smem;                 // [4][64][64] (aliases xh/fh)
    // Region B (80KB): projected tensors (f16). Qs reused for attention output Os.
    _Float16* Qs = (_Float16*)(smem + 32 * 1024);   // [64][128]  (scale pre-folded)
    _Float16* Ks = Qs + NTOK * DIMC;                // [64][128]
    _Float16* Vs = Ks + NTOK * DIMC;                // [64][128]
    _Float16* Qf = Vs + NTOK * DIMC;                // [64][128]  (scale pre-folded)
    _Float16* Kf = Qf + NTOK * DIMC;                // [64][128]
    _Float16* Os = Qs;                              // [64][128]  (aliases Qs)

    const int tid  = threadIdx.x;
    const int wid  = tid >> 5;
    const int lane = tid & 31;
    const int ln   = lane & 15, g = lane >> 4;
    const int b    = blockIdx.x;

    // ---- Phase 1: stage x,f as f16 in LDS ----
    {
        const float4* x4 = (const float4*)(x + (size_t)b * NTOK * DIMC);
        const float4* f4 = (const float4*)(f + (size_t)b * NTOK * DIMC);
        v4h* xo = (v4h*)xh;
        v4h* fo = (v4h*)fh;
        for (int i = tid; i < NTOK * DIMC / 4; i += 256) {
            float4 a = x4[i];
            v4h ha; ha[0] = (_Float16)a.x; ha[1] = (_Float16)a.y;
                    ha[2] = (_Float16)a.z; ha[3] = (_Float16)a.w;
            xo[i] = ha;
            float4 c = f4[i];
            v4h hc; hc[0] = (_Float16)c.x; hc[1] = (_Float16)c.y;
                    hc[2] = (_Float16)c.z; hc[3] = (_Float16)c.w;
            fo[i] = hc;
        }
    }
    __syncthreads();

    // ---- Phase 2: fused projections: Q,K,V from x ; Qf,Kf from f ----
    // 160 output tiles of 16x16, K-loop 4 x 32. Weight fragments: one 32B load from d_ws.
    for (int t = wid; t < 160; t += 8) {
        const int mt = t & 3;
        const int ct = t >> 2;  // 0..39 column tile of the virtual concat [Q|K|V|Qf|Kf]
        const _Float16* A; const _Float16* Wswz; const float* bias;
        _Float16* dst; int nt, dcol, bidx; float scl;
        if (ct < 8)       { A = xh; Wswz = ws + WS_WQ;   bias = bq;    dst = Qs; dcol = ct * 16;        nt = ct;            bidx = dcol;        scl = QKSCALE; }
        else if (ct < 16) { A = xh; Wswz = ws + WS_WKV;  bias = bkv;   dst = Ks; dcol = (ct - 8) * 16;  nt = ct - 8;        bidx = dcol;        scl = 1.0f; }
        else if (ct < 24) { A = xh; Wswz = ws + WS_WKV;  bias = bkv;   dst = Vs; dcol = (ct - 16) * 16; nt = (ct - 16) + 8; bidx = dcol + DIMC; scl = 1.0f; }
        else if (ct < 32) { A = fh; Wswz = ws + WS_WQF;  bias = bq_f;  dst = Qf; dcol = (ct - 24) * 16; nt = ct - 24;       bidx = dcol;        scl = QKSCALE; }
        else              { A = fh; Wswz = ws + WS_WKVF; bias = bkv_f; dst = Kf; dcol = (ct - 32) * 16; nt = ct - 32;       bidx = dcol;        scl = 1.0f; }
        v8f acc = {};
#pragma unroll
        for (int ks = 0; ks < 4; ++ks) {
            v16h af = load_frag_rowmajor(A, mt * 16, ks * 32, DIMC);
            v16h bf = load_fragB_swz(Wswz, nt, ks);
            acc = wmma_f16(af, bf, acc);
        }
        const float bv = bias[bidx + ln];
#pragma unroll
        for (int j = 0; j < 8; ++j) {
            const int row = mt * 16 + j + 8 * g;
            dst[row * DIMC + dcol + ln] = (_Float16)((acc[j] + bv) * scl);
        }
    }
    __syncthreads();

    // ---- Phase 3: S = Q K^T, Sf = Qf Kf^T, fuse rel-pos bias and joint product ----
    for (int t = wid; t < 16; t += 8) {
        const int h = t >> 2, mt = t & 3;
        v16h aq  = load_frag_rowmajor(Qs, mt * 16, h * HD, DIMC);
        v16h aqf = load_frag_rowmajor(Qf, mt * 16, h * HD, DIMC);
#pragma unroll
        for (int nt = 0; nt < 4; ++nt) {
            // B^T (= K rows) is row-major in LDS -> same loader as A.
            v16h bk  = load_frag_rowmajor(Ks, nt * 16, h * HD, DIMC);
            v16h bkf = load_frag_rowmajor(Kf, nt * 16, h * HD, DIMC);
            v8f z0 = {}; v8f z1 = {};
            v8f s  = wmma_f16(aq,  bk,  z0);
            v8f sf = wmma_f16(aqf, bkf, z1);
            const int m = nt * 16 + ln;  // key token
#pragma unroll
            for (int j = 0; j < 8; ++j) {
                const int n  = mt * 16 + j + 8 * g;  // query token
                const int dh = (n >> 3) - (m >> 3) + 7;
                const int dw = (n & 7)  - (m & 7)  + 7;
                const float bv = bias_table[(dh * 15 + dw) * HEADS + h];
                const float comb = (s[j] + bv) * (sf[j] + bv);
                SP[(h * NTOK + n) * NTOK + m] = (_Float16)comb;
            }
        }
    }
    __syncthreads();

    // ---- Phase 4: softmax over keys, one thread per (head,row), in place ----
    {
        _Float16* row = SP + tid * NTOK;
        float mx = -3.0e38f;
        for (int i = 0; i < NTOK; ++i) mx = fmaxf(mx, (float)row[i]);
        float sum = 0.0f;
        for (int i = 0; i < NTOK; ++i) sum += __expf((float)row[i] - mx);
        const float inv = 1.0f / sum;
        for (int i = 0; i < NTOK; ++i)
            row[i] = (_Float16)(__expf((float)row[i] - mx) * inv);
    }
    __syncthreads();

    // ---- Phase 5: O = P @ V  (per head 64x64 @ 64x32) ----
    for (int t = wid; t < 32; t += 8) {
        const int h = t >> 3, mt = (t >> 1) & 3, dt = t & 1;
        v8f acc = {};
#pragma unroll
        for (int ks = 0; ks < 2; ++ks) {
            v16h ap = load_frag_rowmajor(SP + h * NTOK * NTOK, mt * 16, ks * 32, NTOK);
            v16h bv = load_fragB_colsrc(Vs, ks * 32, h * HD + dt * 16, DIMC);
            acc = wmma_f16(ap, bv, acc);
        }
#pragma unroll
        for (int j = 0; j < 8; ++j) {
            const int row = mt * 16 + j + 8 * g;
            Os[row * DIMC + h * HD + dt * 16 + ln] = (_Float16)acc[j];
        }
    }
    __syncthreads();

    // ---- Phase 6: out = O @ wproj + bproj (fp32 store) ----
    for (int t = wid; t < 32; t += 8) {
        const int mt = t & 3, nc = t >> 2;
        v8f acc = {};
#pragma unroll
        for (int ks = 0; ks < 4; ++ks) {
            v16h ao = load_frag_rowmajor(Os, mt * 16, ks * 32, DIMC);
            v16h bw = load_fragB_swz(ws + WS_WPROJ, nc, ks);
            acc = wmma_f16(ao, bw, acc);
        }
        const int col = nc * 16 + ln;
        const float bv = bproj[col];
        float* og = out + (size_t)b * NTOK * DIMC;
#pragma unroll
        for (int j = 0; j < 8; ++j) {
            const int row = mt * 16 + j + 8 * g;
            og[row * DIMC + col] = acc[j] + bv;
        }
    }
}

extern "C" void kernel_launch(void* const* d_in, const int* in_sizes, int n_in,
                              void* d_out, int out_size, void* d_ws, size_t ws_size,
                              hipStream_t stream) {
    const float* x      = (const float*)d_in[0];
    const float* f      = (const float*)d_in[1];
    const float* wq     = (const float*)d_in[2];
    const float* bq     = (const float*)d_in[3];
    const float* wkv    = (const float*)d_in[4];
    const float* bkv    = (const float*)d_in[5];
    const float* wq_f   = (const float*)d_in[6];
    const float* bq_f   = (const float*)d_in[7];
    const float* wkv_f  = (const float*)d_in[8];
    const float* bkv_f  = (const float*)d_in[9];
    const float* btab   = (const float*)d_in[10];
    const float* wproj  = (const float*)d_in[11];
    const float* bproj  = (const float*)d_in[12];
    float* out = (float*)d_out;
    _Float16* ws = (_Float16*)d_ws;

    const int nwin = in_sizes[0] / (NTOK * DIMC);   // 4096
    const size_t shmem = 112u * 1024u;              // dynamic LDS (WGP has 320KB)

    (void)hipFuncSetAttribute((const void*)win_joint_attn_kernel,
                              hipFuncAttributeMaxDynamicSharedMemorySize, (int)shmem);

    // Stream-ordered: swizzle weights into d_ws, then run attention.
    swizzle_weights_kernel<<<dim3((WS_TOTAL + 255) / 256), dim3(256), 0, stream>>>(
        wq, wkv, wq_f, wkv_f, wproj, ws);

    win_joint_attn_kernel<<<dim3(nwin), dim3(256), shmem, stream>>>(
        x, f, bq, bkv, bq_f, bkv_f, btab, bproj, ws, out);
}
